// TransducerModel_84361747628192
// MI455X (gfx1250) — matrix operand order repaired and checked by reading, not compiled, not run on hardware
//
#include <hip/hip_runtime.h>
#include <hip/hip_bf16.h>
#include <stdint.h>

typedef __attribute__((ext_vector_type(16))) _Float16 v16h;
typedef __attribute__((ext_vector_type(8)))  _Float16 v8h;
typedef __attribute__((ext_vector_type(8)))  float    v8f;

#define WMMA_F16(a, b, c) \
  __builtin_amdgcn_wmma_f32_16x16x32_f16(false, (a), false, (b), (short)0, (c), false, false)

__device__ __forceinline__ v16h cat8(v8h lo, v8h hi) {
  return __builtin_shufflevector(lo, hi, 0, 1, 2, 3, 4, 5, 6, 7, 8, 9, 10, 11, 12, 13, 14, 15);
}
__device__ __forceinline__ v16h zero16() {
  v16h z;
#pragma unroll
  for (int i = 0; i < 16; ++i) z[i] = (_Float16)0.f;
  return z;
}
// halves p[0..7] and p[16..23] (16B-aligned p) -> one A/B fragment (two b128 loads)
__device__ __forceinline__ v16h ldfrag(const _Float16* p) {
  return cat8(*(const v8h*)p, *(const v8h*)(p + 16));
}

// ---------------------------------------------------------------------------
// WMMA GEMM: C[M,N](f32 and/or f16-transposed) = A(f16,[M,K]) @ Bt(f16,[Np,K]) + bias
// Bt is TRANSPOSED and zero-padded to Np = roundup64(N): B-fragment loads are
// two b128 per fragment with constant offsets, no guards.
// Block = 128 thr (4 waves); each wave: 1 M-tile x 4 N-tiles (4 indep WMMA chains).
// ---------------------------------------------------------------------------
__global__ void wmma_gemm(const _Float16* __restrict__ A, const _Float16* __restrict__ Bt,
                          const float* __restrict__ bias, float* __restrict__ Cf,
                          _Float16* __restrict__ ChT,
                          int M, int N, int K, int lda, int ldc, int ldct) {
  const int lane = threadIdx.x & 31;
  const int w = threadIdx.x >> 5;
  const int mrow = (blockIdx.y * 4 + w) * 16;
  const int ncol0 = blockIdx.x * 64;
  if (mrow >= M) return;  // uniform per wave

  const int am = mrow + (lane & 15);
  const int koff = (lane >> 4) * 8;   // A: lanes 0-15 K=0..7/16..23, lanes 16-31 K=8..15/24..31
  const int kk = (lane >> 4) * 8;     // B: lanes 0-15 K=0..7/16..23, lanes 16-31 K=8..15/24..31

  const _Float16* ap = A + (size_t)(am < M ? am : 0) * lda + koff;
  const _Float16* bp0 = Bt + (size_t)(ncol0 + (lane & 15)) * K + kk;
  const size_t tstep = (size_t)16 * K;  // between N-tiles

  v8f acc[4] = {{}, {}, {}, {}};
  if (mrow + 16 <= M) {  // common fast path: no A guards
    for (int kb = 0; kb < K; kb += 32) {
      v16h a = ldfrag(ap + kb);
      v16h b0 = ldfrag(bp0 + kb);
      v16h b1 = ldfrag(bp0 + tstep + kb);
      v16h b2 = ldfrag(bp0 + 2 * tstep + kb);
      v16h b3 = ldfrag(bp0 + 3 * tstep + kb);
      acc[0] = WMMA_F16(a, b0, acc[0]);
      acc[1] = WMMA_F16(a, b1, acc[1]);
      acc[2] = WMMA_F16(a, b2, acc[2]);
      acc[3] = WMMA_F16(a, b3, acc[3]);
    }
  } else {               // M-edge tile: per-lane guarded A
    const bool aok = (am < M);
    for (int kb = 0; kb < K; kb += 32) {
      v16h a = aok ? ldfrag(ap + kb) : zero16();
      v16h b0 = ldfrag(bp0 + kb);
      v16h b1 = ldfrag(bp0 + tstep + kb);
      v16h b2 = ldfrag(bp0 + 2 * tstep + kb);
      v16h b3 = ldfrag(bp0 + 3 * tstep + kb);
      acc[0] = WMMA_F16(a, b0, acc[0]);
      acc[1] = WMMA_F16(a, b1, acc[1]);
      acc[2] = WMMA_F16(a, b2, acc[2]);
      acc[3] = WMMA_F16(a, b3, acc[3]);
    }
  }

  const int nl = lane & 15;
  const int mb = mrow + (lane >> 4) * 8;
#pragma unroll
  for (int j = 0; j < 4; ++j) {
    int n = ncol0 + 16 * j + nl;
    if (n < N) {
      float bv = (bias != nullptr) ? bias[n] : 0.f;
#pragma unroll
      for (int i = 0; i < 8; ++i) {
        int m = mb + i;
        if (m < M) {
          float v = acc[j][i] + bv;
          if (Cf) Cf[(size_t)m * ldc + n] = v;
          if (ChT) ChT[(size_t)n * ldct + m] = (_Float16)v;
        }
      }
    }
  }
}

// ---------------------------------------------------------------------------
// Persistent GRU recurrence. grid.x = direction (0=fwd, 1=bwd-time).
// Block = 512 thr = 16 waves; wave w owns cols [16w,16w+16) and gate tiles
// {w, w+16, w+32} of rec = h @ Wr (WrT transposed [768,256] f16).
// h: f16 ping-pong in LDS (WMMA A), f32 in registers (lanes 0-15 x 8 batch).
// All steady-state loads use constant offsets from hoisted base pointers.
// ---------------------------------------------------------------------------
__device__ __forceinline__ float sigmf(float x) { return 1.f / (1.f + __expf(-x)); }

__global__ void gru_rec(const float* __restrict__ preA, const float* __restrict__ preB,
                        const _Float16* __restrict__ WrA, const _Float16* __restrict__ WrB,
                        const float* __restrict__ brA, const float* __restrict__ brB,
                        const float* __restrict__ h0vec, _Float16* __restrict__ out,
                        int S, int ostride, int ooffA, int ooffB) {
  __shared__ _Float16 h16[2][16][256];  // 16 KB, rows 8-15 stay zero (M pad)

  const int dir = blockIdx.x;
  const float* pre   = dir ? preB : preA;
  const _Float16* Wr = dir ? WrB : WrA;
  const float* brec  = dir ? brB : brA;
  const int ooff     = dir ? ooffB : ooffA;
  const bool rev     = (dir == 1);

  const int tid = threadIdx.x;
  const int lane = tid & 31;
  const int w = tid >> 5;
  const int col = 16 * w + (lane & 15);
  const int koff = (lane >> 4) * 8;
  const int kk = (lane >> 4) * 8;

  // hoisted per-lane base pointers (constant offsets inside the t-loop)
  const _Float16* wz = Wr + (size_t)(col) * 256 + kk;
  const _Float16* wr = Wr + (size_t)(256 + col) * 256 + kk;
  const _Float16* wh = Wr + (size_t)(512 + col) * 256 + kk;
  const _Float16* a0 = &h16[0][lane & 15][koff];
  const _Float16* a1 = &h16[1][lane & 15][koff];

  for (int i = tid; i < 2 * 16 * 256; i += blockDim.x)
    ((_Float16*)h16)[i] = (_Float16)0.f;

  float hreg[8];
#pragma unroll
  for (int i = 0; i < 8; ++i) hreg[i] = (h0vec != nullptr) ? h0vec[col] : 0.f;
  __syncthreads();
  if (lane < 16) {
#pragma unroll
    for (int b = 0; b < 8; ++b) h16[0][b][col] = (_Float16)hreg[b];
  }
  __syncthreads();

  int cur = 0;
  for (int t = 0; t < S; ++t) {
    const int sidx = rev ? (S - 1 - t) : t;
    const _Float16* ab = cur ? a1 : a0;
    v8f cz = {}, cr = {}, ch = {};
#pragma unroll
    for (int kc = 0; kc < 256; kc += 32) {
      v16h a  = ldfrag(ab + kc);
      v16h bz = ldfrag(wz + kc);
      v16h br = ldfrag(wr + kc);
      v16h bh = ldfrag(wh + kc);
      cz = WMMA_F16(a, bz, cz);
      cr = WMMA_F16(a, br, cr);
      ch = WMMA_F16(a, bh, ch);
    }
    const int nxt = cur ^ 1;
    if (lane < 16) {
#pragma unroll
      for (int b = 0; b < 8; ++b) {
        const float* pr = pre + ((size_t)b * S + sidx) * 768;
        float z  = sigmf(pr[col]       + cz[b] + brec[col]);
        float r  = sigmf(pr[256 + col] + cr[b] + brec[256 + col]);
        float hh = tanhf(pr[512 + col] + r * (ch[b] + brec[512 + col]));
        float hn = z * hreg[b] + (1.f - z) * hh;
        hreg[b] = hn;
        _Float16 hv = (_Float16)hn;
        h16[nxt][b][col] = hv;
        out[((size_t)b * S + sidx) * ostride + ooff + col] = hv;
      }
    }
    __syncthreads();
    cur = nxt;
  }
}

// ---------------------------------------------------------------------------
// Small helpers
// ---------------------------------------------------------------------------
__global__ void cvt_f32_f16(const float* __restrict__ src, _Float16* __restrict__ dst, int n) {
  int i = blockIdx.x * blockDim.x + threadIdx.x;
  if (i < n) dst[i] = (_Float16)src[i];
}

// src [K,N] f32 row-major -> dst [Np,K] f16 (transposed, rows n>=N zeroed)
__global__ void cvt_transpose(const float* __restrict__ src, _Float16* __restrict__ dst,
                              int K, int N, int Np) {
  int i = blockIdx.x * blockDim.x + threadIdx.x;
  if (i >= K * Np) return;
  int k = i % K, n = i / K;
  dst[i] = (n < N) ? (_Float16)src[(size_t)k * N + n] : (_Float16)0.f;
}

__global__ void zero_f16(_Float16* __restrict__ p, int n) {
  int i = blockIdx.x * blockDim.x + threadIdx.x;
  if (i < n) p[i] = (_Float16)0.f;
}

__global__ void embed_enc_kernel(const int* __restrict__ tok, const float* __restrict__ emb,
                                 _Float16* __restrict__ out) {
  int row = blockIdx.x, d = threadIdx.x;           // row in [0, B*T)
  out[(size_t)row * 256 + d] = (_Float16)emb[(size_t)tok[row] * 256 + d];
}

__global__ void embed_pred_kernel(const int* __restrict__ y, const float* __restrict__ emb,
                                  _Float16* __restrict__ out) {
  int row = blockIdx.x, d = threadIdx.x;           // row in [0, B*129)
  int b = row / 129, u = row % 129;
  int t = (u == 0) ? 0 : y[b * 128 + (u - 1)];     // NULL_INDEX prepended
  out[(size_t)row * 256 + d] = (_Float16)emb[(size_t)t * 256 + d];
}

__global__ void matvec_bias(const float* __restrict__ vec, const float* __restrict__ W,
                            float* __restrict__ outv, int K, int N) {
  int v = blockIdx.x * blockDim.x + threadIdx.x;
  if (v >= N) return;
  float s = 0.f;
  for (int k = 0; k < K; ++k) s += vec[k] * W[(size_t)k * N + v];
  outv[v] = s;
}

__global__ void joiner_add(const float* __restrict__ fe, const float* __restrict__ fp,
                           const float* __restrict__ jb, float* __restrict__ out, int total) {
  int idx = blockIdx.x * blockDim.x + threadIdx.x;
  if (idx >= total) return;
  int v = idx % 65;
  int r = idx / 65;
  int u = r % 129;
  int r2 = r / 129;
  int t = r2 % 256;
  int b = r2 / 256;
  out[idx] = fe[((size_t)b * 256 + t) * 65 + v] + fp[((size_t)b * 129 + u) * 65 + v] + jb[v];
}

// ---------------------------------------------------------------------------
// Host launcher
// ---------------------------------------------------------------------------
extern "C" void kernel_launch(void* const* d_in, const int* in_sizes, int n_in,
                              void* d_out, int out_size, void* d_ws, size_t ws_size,
                              hipStream_t stream) {
  (void)in_sizes; (void)n_in; (void)out_size; (void)ws_size;
  const int* x        = (const int*)d_in[0];
  const int* y        = (const int*)d_in[1];
  const float* emb_enc = (const float*)d_in[4];
  const float* k1f = (const float*)d_in[5],  *r1f = (const float*)d_in[6],  *b1f = (const float*)d_in[7];
  const float* k1b = (const float*)d_in[8],  *r1b = (const float*)d_in[9],  *b1b = (const float*)d_in[10];
  const float* k2f = (const float*)d_in[11], *r2f = (const float*)d_in[12], *b2f = (const float*)d_in[13];
  const float* k2b = (const float*)d_in[14], *r2b = (const float*)d_in[15], *b2b = (const float*)d_in[16];
  const float* k3f = (const float*)d_in[17], *r3f = (const float*)d_in[18], *b3f = (const float*)d_in[19];
  const float* k3b = (const float*)d_in[20], *r3b = (const float*)d_in[21], *b3b = (const float*)d_in[22];
  const float* enc_w = (const float*)d_in[23], *enc_b = (const float*)d_in[24];
  const float* emb_pred = (const float*)d_in[25];
  const float* pk = (const float*)d_in[26], *pr = (const float*)d_in[27], *pb = (const float*)d_in[28];
  const float* pred_w = (const float*)d_in[29], *pred_b = (const float*)d_in[30];
  const float* h0  = (const float*)d_in[31];
  const float* j_w = (const float*)d_in[32], *j_b = (const float*)d_in[33];

  // ---- bump allocator over workspace (deterministic, recomputed each call)
  char* wsp = (char*)d_ws;
  size_t pos = 0;
  auto alloc = [&](size_t bytes) -> void* {
    void* p = wsp + pos;
    pos += (bytes + 255) & ~(size_t)255;
    return p;
  };
  auto h16a = [&](size_t n) { return (_Float16*)alloc(n * sizeof(_Float16)); };
  auto f32a = [&](size_t n) { return (float*)alloc(n * sizeof(float)); };

  // transposed B-operand weights [Np, K]
  _Float16 *k1fT = h16a(768 * 256), *r1fT = h16a(768 * 256);
  _Float16 *k1bT = h16a(768 * 256), *r1bT = h16a(768 * 256);
  _Float16 *k2fT = h16a(768 * 512), *r2fT = h16a(768 * 256);
  _Float16 *k2bT = h16a(768 * 512), *r2bT = h16a(768 * 256);
  _Float16 *k3fT = h16a(768 * 512), *r3fT = h16a(768 * 256);
  _Float16 *k3bT = h16a(768 * 512), *r3bT = h16a(768 * 256);
  _Float16 *pkT = h16a(768 * 256), *prT = h16a(768 * 256);
  _Float16 *jwT = h16a(128 * 256);                 // N=65 -> Np=128
  _Float16 *wfeT = h16a(128 * 512), *wfpT = h16a(128 * 256);
  // A-operand weights (row-major)
  _Float16 *encw16 = h16a(512 * 256), *predw16 = h16a(256 * 256);
  // activations
  _Float16 *e0 = h16a(2048 * 256);
  _Float16 *enc1 = h16a(2048 * 512), *enc2 = h16a(2048 * 512), *enc3 = h16a(2048 * 512);
  _Float16 *gin = h16a(1032 * 256), *g = h16a(1032 * 256);
  float *preF = f32a(2048 * 768), *preB = f32a(2048 * 768), *ppre = f32a(1032 * 768);
  float *fe = f32a(2048 * 65), *fp = f32a(1032 * 65);
  float *bfe = f32a(65), *bfp = f32a(65);

  auto cvt = [&](const float* s, _Float16* d, int n) {
    cvt_f32_f16<<<(n + 255) / 256, 256, 0, stream>>>(s, d, n);
  };
  auto cvtT = [&](const float* s, _Float16* d, int K, int N, int Np) {
    cvt_transpose<<<(K * Np + 255) / 256, 256, 0, stream>>>(s, d, K, N, Np);
  };
  auto gemm = [&](const _Float16* A, const _Float16* Bt, const float* bias,
                  float* Cf, _Float16* ChT, int M, int N, int K, int lda, int ldct) {
    dim3 grid((N + 63) / 64, (M + 63) / 64);
    wmma_gemm<<<grid, 128, 0, stream>>>(A, Bt, bias, Cf, ChT, M, N, K, lda, N, ldct);
  };

  // ---- weight conversion/transpose
  cvtT(k1f, k1fT, 256, 768, 768); cvtT(r1f, r1fT, 256, 768, 768);
  cvtT(k1b, k1bT, 256, 768, 768); cvtT(r1b, r1bT, 256, 768, 768);
  cvtT(k2f, k2fT, 512, 768, 768); cvtT(r2f, r2fT, 256, 768, 768);
  cvtT(k2b, k2bT, 512, 768, 768); cvtT(r2b, r2bT, 256, 768, 768);
  cvtT(k3f, k3fT, 512, 768, 768); cvtT(r3f, r3fT, 256, 768, 768);
  cvtT(k3b, k3bT, 512, 768, 768); cvtT(r3b, r3bT, 256, 768, 768);
  cvtT(pk, pkT, 256, 768, 768);   cvtT(pr, prT, 256, 768, 768);
  cvtT(j_w, jwT, 256, 65, 128);
  cvt(enc_w, encw16, 512 * 256);  cvt(pred_w, predw16, 256 * 256);

  // ---- fuse joiner: WfeT = (enc_w @ j_w)^T, WfpT = (pred_w @ j_w)^T ; fused biases
  zero_f16<<<(128 * 512 + 255) / 256, 256, 0, stream>>>(wfeT, 128 * 512);
  zero_f16<<<(128 * 256 + 255) / 256, 256, 0, stream>>>(wfpT, 128 * 256);
  gemm(encw16, jwT, nullptr, nullptr, wfeT, 512, 65, 256, 256, 512);
  gemm(predw16, jwT, nullptr, nullptr, wfpT, 256, 65, 256, 256, 256);
  matvec_bias<<<1, 128, 0, stream>>>(enc_b, j_w, bfe, 256, 65);
  matvec_bias<<<1, 128, 0, stream>>>(pred_b, j_w, bfp, 256, 65);

  // ---- encoder
  embed_enc_kernel<<<2048, 256, 0, stream>>>(x, emb_enc, e0);

  gemm(e0, k1fT, b1f, preF, nullptr, 2048, 768, 256, 256, 0);
  gemm(e0, k1bT, b1b, preB, nullptr, 2048, 768, 256, 256, 0);
  gru_rec<<<2, 512, 0, stream>>>(preF, preB, r1fT, r1bT, b1f + 768, b1b + 768,
                                 nullptr, enc1, 256, 512, 0, 256);

  gemm(enc1, k2fT, b2f, preF, nullptr, 2048, 768, 512, 512, 0);
  gemm(enc1, k2bT, b2b, preB, nullptr, 2048, 768, 512, 512, 0);
  gru_rec<<<2, 512, 0, stream>>>(preF, preB, r2fT, r2bT, b2f + 768, b2b + 768,
                                 nullptr, enc2, 256, 512, 0, 256);

  gemm(enc2, k3fT, b3f, preF, nullptr, 2048, 768, 512, 512, 0);
  gemm(enc2, k3bT, b3b, preB, nullptr, 2048, 768, 512, 512, 0);
  gru_rec<<<2, 512, 0, stream>>>(preF, preB, r3fT, r3bT, b3f + 768, b3b + 768,
                                 nullptr, enc3, 256, 512, 0, 256);

  // ---- predictor
  embed_pred_kernel<<<1032, 256, 0, stream>>>(y, emb_pred, gin);
  gemm(gin, pkT, pb, ppre, nullptr, 1032, 768, 256, 256, 0);
  gru_rec<<<1, 512, 0, stream>>>(ppre, ppre, prT, prT, pb + 768, pb + 768,
                                 h0, g, 129, 256, 0, 0);

  // ---- joiner factors + broadcast add
  gemm(enc3, wfeT, bfe, fe, nullptr, 2048, 65, 512, 512, 0);
  gemm(g, wfpT, bfp, fp, nullptr, 1032, 65, 256, 256, 0);

  const int total = 8 * 256 * 129 * 65;
  joiner_add<<<(total + 255) / 256, 256, 0, stream>>>(fe, fp, j_b, (float*)d_out, total);
}
